// DeformTransformerDecoderLayer_45767171506677
// MI455X (gfx1250) — compile-verified, matrix operand order checked by reading, not run on hardware
//
#include <hip/hip_runtime.h>
#include <hip/hip_bf16.h>
#include <stdint.h>
#include <math.h>

// ---------------------------------------------------------------------------
// Types for CDNA5 WMMA (wave32, 16x16x32 bf16 -> f32)
// ---------------------------------------------------------------------------
typedef __attribute__((ext_vector_type(16))) __bf16 v16bf;
typedef __attribute__((ext_vector_type(8)))  float  v8f;

#define TB 256
#define BM 128   // block tile M (4 waves x 32 rows)
#define BN 128   // block tile N (2 waves x 64 cols)
#define BK 32    // K step = WMMA K
#define LDSPAD 40  // ushorts per LDS row (32 data + 8 pad), keeps 16B alignment

__device__ __forceinline__ unsigned short f2bf(float f) {
  unsigned int u = __float_as_uint(f);
  u += 0x7FFFu + ((u >> 16) & 1u);          // round-to-nearest-even
  return (unsigned short)(u >> 16);
}
__device__ __forceinline__ float bf2f(unsigned short s) {
  return __uint_as_float(((unsigned int)s) << 16);
}

// ---------------------------------------------------------------------------
// Generic batched strided GEMM:  C = alpha * (A @ B) + bias  (optional ReLU,
// optional bf16 output).  A,B are fp32 in global with arbitrary strides,
// converted to bf16 while staging into LDS; math runs on v_wmma.
// Each wave computes a 32x64 C region = 2x4 WMMA tiles: 8 WMMAs per 6
// fragment fetches per K step.  Staging iteration order is chosen from the
// unit stride so global loads stay lane-contiguous for every layout.
// Batch index z decomposes as (z / zdiv, z % zdiv) with two stride pairs.
// M must be a multiple of 128; N is column-guarded; K a multiple of 32.
// ---------------------------------------------------------------------------
struct GemmParams {
  const float* A; const float* B; const float* bias; void* C;
  int M, N, K;
  long long sa_m, sa_k, sb_k, sb_n, sc_m, sc_n;
  long long aoff0, boff0, coff0;
  int zdiv;
  long long a_b1, a_b2, b_b1, b_b2, c_b1, c_b2;
  float alpha;
  int relu, out_bf16;
};

__global__ __launch_bounds__(TB) void wmma_gemm_bf16(GemmParams p) {
  __shared__ unsigned short As[BM][LDSPAD];   // A tile, row-major [m][k]
  __shared__ unsigned short Bs[BN][LDSPAD];   // B tile transposed [n][k]

  const int z  = blockIdx.z;
  const int zq = z / p.zdiv, zr = z % p.zdiv;
  const long long ao = p.aoff0 + (long long)zq * p.a_b1 + (long long)zr * p.a_b2;
  const long long bo = p.boff0 + (long long)zq * p.b_b1 + (long long)zr * p.b_b2;
  const long long co = p.coff0 + (long long)zq * p.c_b1 + (long long)zr * p.c_b2;

  const int m0 = blockIdx.x * BM;
  const int n0 = blockIdx.y * BN;
  const int tid  = threadIdx.x;
  const int lane = tid & 31;
  const int wid  = tid >> 5;
  const int wm = wid & 3;    // 4 waves along M (32 rows each)
  const int wn = wid >> 2;   // 2 waves along N (64 cols each)

  v8f acc[2][4];
#pragma unroll
  for (int i = 0; i < 2; ++i)
#pragma unroll
    for (int j = 0; j < 4; ++j)
      acc[i][j] = (v8f){0.f, 0.f, 0.f, 0.f, 0.f, 0.f, 0.f, 0.f};

  union Frag { unsigned int u[8]; v16bf v; };

  const bool a_kfast = (p.sa_k == 1);
  const bool b_kfast = (p.sb_k == 1);

  for (int k0 = 0; k0 < p.K; k0 += BK) {
    // ---- stage A tile (128x32) as bf16, lane-contiguous along unit stride
    if (a_kfast) {
      for (int i = tid; i < BM * BK; i += TB) {
        int m = i >> 5, k = i & 31;
        As[m][k] = f2bf(p.A[ao + (long long)(m0 + m) * p.sa_m +
                                 (long long)(k0 + k) * p.sa_k]);
      }
    } else {
      for (int i = tid; i < BM * BK; i += TB) {
        int m = i & (BM - 1), k = i >> 7;
        As[m][k] = f2bf(p.A[ao + (long long)(m0 + m) * p.sa_m +
                                 (long long)(k0 + k) * p.sa_k]);
      }
    }
    // ---- stage B tile (128n x 32k, n-major) as bf16, with column guard
    if (b_kfast) {
      for (int i = tid; i < BN * BK; i += TB) {
        int n = i >> 5, k = i & 31;
        float f = (n0 + n < p.N)
                      ? p.B[bo + (long long)(k0 + k) * p.sb_k +
                                (long long)(n0 + n) * p.sb_n]
                      : 0.f;
        Bs[n][k] = f2bf(f);
      }
    } else {
      for (int i = tid; i < BN * BK; i += TB) {
        int n = i & (BN - 1), k = i >> 7;
        float f = (n0 + n < p.N)
                      ? p.B[bo + (long long)(k0 + k) * p.sb_k +
                                (long long)(n0 + n) * p.sb_n]
                      : 0.f;
        Bs[n][k] = f2bf(f);
      }
    }
    __syncthreads();

    // Assemble fragments per ISA 16-bit A/B layout:
    // lanes 0-15 hold K = {0..7, 16..23}, lanes 16-31 hold K = {8..15, 24..31}
    const int kh = lane >> 4;
    const int ll = lane & 15;
    Frag a[2], b[4];
#pragma unroll
    for (int s = 0; s < 2; ++s) {
      const unsigned int* Ar = (const unsigned int*)(&As[wm * 32 + s * 16 + ll][0]);
#pragma unroll
      for (int j = 0; j < 4; ++j) {
        a[s].u[j]     = Ar[kh * 4 + j];
        a[s].u[4 + j] = Ar[8 + kh * 4 + j];
      }
    }
#pragma unroll
    for (int s = 0; s < 4; ++s) {
      const unsigned int* Br = (const unsigned int*)(&Bs[wn * 64 + s * 16 + ll][0]);
#pragma unroll
      for (int j = 0; j < 4; ++j) {
        b[s].u[j]     = Br[kh * 4 + j];
        b[s].u[4 + j] = Br[8 + kh * 4 + j];
      }
    }
#pragma unroll
    for (int i = 0; i < 2; ++i)
#pragma unroll
      for (int j = 0; j < 4; ++j)
        acc[i][j] = __builtin_amdgcn_wmma_f32_16x16x32_bf16(
            false, a[i].v, false, b[j].v, (short)0, acc[i][j], false, false);
    __syncthreads();
  }

  // Epilogue. C/D layout: VGPR j, lanes 0-15 -> M=j, lanes 16-31 -> M=8+j.
  const int lm = (lane >> 4) ? 8 : 0;
  const int ln = lane & 15;
#pragma unroll
  for (int sm = 0; sm < 2; ++sm) {
#pragma unroll
    for (int sn = 0; sn < 4; ++sn) {
      const int cn = n0 + wn * 64 + sn * 16 + ln;
      if (cn >= p.N) continue;
      const int cm0 = m0 + wm * 32 + sm * 16 + lm;
      const float bv = p.bias ? p.bias[cn] : 0.f;
      const v8f& av = acc[sm][sn];
#pragma unroll
      for (int j = 0; j < 8; ++j) {
        float v = p.alpha * av[j] + bv;
        if (p.relu) v = fmaxf(v, 0.f);
        long long ci =
            co + (long long)(cm0 + j) * p.sc_m + (long long)cn * p.sc_n;
        if (p.out_bf16) ((unsigned short*)p.C)[ci] = f2bf(v);
        else            ((float*)p.C)[ci] = v;
      }
    }
  }
}

// ---------------------------------------------------------------------------
// Row softmax over contiguous rows of length rowlen (attention scores).
// ---------------------------------------------------------------------------
__global__ __launch_bounds__(TB) void row_softmax(float* x, int rowlen) {
  __shared__ float red[TB];
  float* p = x + (long long)blockIdx.x * rowlen;
  const int tid = threadIdx.x;

  float mx = -3.4e38f;
  for (int i = tid; i < rowlen; i += TB) mx = fmaxf(mx, p[i]);
  red[tid] = mx; __syncthreads();
  for (int s = TB / 2; s > 0; s >>= 1) {
    if (tid < s) red[tid] = fmaxf(red[tid], red[tid + s]);
    __syncthreads();
  }
  mx = red[0]; __syncthreads();

  float sum = 0.f;
  for (int i = tid; i < rowlen; i += TB) {
    float e = __expf(p[i] - mx);
    p[i] = e;
    sum += e;
  }
  red[tid] = sum; __syncthreads();
  for (int s = TB / 2; s > 0; s >>= 1) {
    if (tid < s) red[tid] += red[tid + s];
    __syncthreads();
  }
  float inv = 1.0f / red[0];
  for (int i = tid; i < rowlen; i += TB) p[i] *= inv;
}

// Softmax over the 12 (L*P) deformable weights per (b, n, head).
// aw layout: [B*N][96], group base = head*12.
__global__ void softmax12(float* aw, int total) {
  int t = blockIdx.x * blockDim.x + threadIdx.x;
  if (t >= total) return;
  int row = t >> 3, hh = t & 7;
  float* p = aw + (long long)row * 96 + hh * 12;
  float mx = p[0];
#pragma unroll
  for (int i = 1; i < 12; ++i) mx = fmaxf(mx, p[i]);
  float sum = 0.f;
  float e[12];
#pragma unroll
  for (int i = 0; i < 12; ++i) { e[i] = __expf(p[i] - mx); sum += e[i]; }
  float inv = 1.0f / sum;
#pragma unroll
  for (int i = 0; i < 12; ++i) p[i] = e[i] * inv;
}

// ---------------------------------------------------------------------------
// Deformable bilinear gather.  One wave per (b, n, head); lane = dh channel.
// values are bf16 [B, H*W, 256]; grid_sample(align_corners=False, zeros).
// ---------------------------------------------------------------------------
__device__ __forceinline__ float samp_corner(const unsigned short* V, int W,
                                             int H, int xi, int yi, float w) {
  if (xi < 0 || xi >= W || yi < 0 || yi >= H) return 0.f;
  return w * bf2f(V[((long long)yi * W + xi) * 256]);
}

__global__ __launch_bounds__(TB) void deform_gather(
    const float* __restrict__ query_pos, const float* __restrict__ off,
    const float* __restrict__ aw, const unsigned short* __restrict__ v0,
    const unsigned short* __restrict__ v1, const unsigned short* __restrict__ v2,
    float* __restrict__ out) {
  const int gw   = (blockIdx.x * TB + threadIdx.x) >> 5;  // global wave id
  const int lane = threadIdx.x & 31;
  const int hh = gw & 7;
  const int n  = (gw >> 3) & 1023;
  const int b  = gw >> 13;
  const int row = b * 1024 + n;

  const float rx = query_pos[(long long)row * 2 + 0] * (1.0f / 352.0f);
  const float ry = query_pos[(long long)row * 2 + 1] * (1.0f / 400.0f);

  const int Hs[3] = {200, 200, 400};
  const int Ws[3] = {176, 176, 352};
  const unsigned short* vals[3] = {v0, v1, v2};

  float acc = 0.f;
#pragma unroll
  for (int l = 0; l < 3; ++l) {
    const int H = Hs[l], W = Ws[l];
    const unsigned short* V =
        vals[l] + (long long)b * (long long)H * W * 256 + hh * 32 + lane;
    const float invW = 1.0f / (float)W, invH = 1.0f / (float)H;
#pragma unroll
    for (int pp = 0; pp < 4; ++pp) {
      const long long ob = (long long)row * 192 + hh * 24 + l * 8 + pp * 2;
      float ox = off[ob + 0], oy = off[ob + 1];
      float w  = aw[(long long)row * 96 + hh * 12 + l * 4 + pp];
      float x = (rx + ox * invW) * (float)W - 0.5f;
      float y = (ry + oy * invH) * (float)H - 0.5f;
      float x0f = floorf(x), y0f = floorf(y);
      int   x0 = (int)x0f, y0 = (int)y0f;
      float wx = x - x0f, wy = y - y0f;
      float s = samp_corner(V, W, H, x0,     y0,     (1.f - wx) * (1.f - wy))
              + samp_corner(V, W, H, x0 + 1, y0,     wx * (1.f - wy))
              + samp_corner(V, W, H, x0,     y0 + 1, (1.f - wx) * wy)
              + samp_corner(V, W, H, x0 + 1, y0 + 1, wx * wy);
      acc += w * s;
    }
  }
  out[(long long)row * 256 + hh * 32 + lane] = acc;
}

// ---------------------------------------------------------------------------
// Residual + LayerNorm over C=256 channels.  Residual and output use
// generalized strides so the [B,C,N] <-> [B,N,C] transposes fold in here.
//   res_idx = (r/rinner)*rb1 + (r%rinner)*rb2 + c*rc
//   out_idx = (r/oinner)*ob1 + (r%oinner)*ob2 + c*oc
// ---------------------------------------------------------------------------
struct LNParams {
  const float* x; const float* res; const float* g; const float* bb;
  float* out;
  int C;
  int rinner; long long rb1, rb2, rc;
  int oinner; long long ob1, ob2, oc;
};

__global__ __launch_bounds__(TB) void ln_residual(LNParams p) {
  __shared__ float s1[TB], s2[TB];
  const int r = blockIdx.x, c = threadIdx.x;
  long long ridx = (long long)(r / p.rinner) * p.rb1 +
                   (long long)(r % p.rinner) * p.rb2 + (long long)c * p.rc;
  float v = p.x[(long long)r * p.C + c] + p.res[ridx];
  s1[c] = v; s2[c] = v * v; __syncthreads();
  for (int s = TB / 2; s > 0; s >>= 1) {
    if (c < s) { s1[c] += s1[c + s]; s2[c] += s2[c + s]; }
    __syncthreads();
  }
  float m   = s1[0] / (float)p.C;
  float var = s2[0] / (float)p.C - m * m;
  float y = (v - m) * rsqrtf(var + 1e-5f) * p.g[c] + p.bb[c];
  long long oidx = (long long)(r / p.oinner) * p.ob1 +
                   (long long)(r % p.oinner) * p.ob2 + (long long)c * p.oc;
  p.out[oidx] = y;
}

// ---------------------------------------------------------------------------
// Host orchestration
// ---------------------------------------------------------------------------
extern "C" void kernel_launch(void* const* d_in, const int* in_sizes, int n_in,
                              void* d_out, int out_size, void* d_ws,
                              size_t ws_size, hipStream_t stream) {
  (void)in_sizes; (void)n_in; (void)out_size; (void)ws_size;
  const float* query = (const float*)d_in[0];   // [2,256,1024]
  const float* qpos  = (const float*)d_in[1];   // [2,1024,2]
  const float* bev   = (const float*)d_in[2];   // [2,256,200,176]
  const float* xc4   = (const float*)d_in[3];   // [2,256,200,176]
  const float* xc3   = (const float*)d_in[4];   // [2,256,400,352]
  const float* ipw   = (const float*)d_in[5];   // [768,256]
  const float* ipb   = (const float*)d_in[6];
  const float* opw   = (const float*)d_in[7];   // [256,256]
  const float* opb   = (const float*)d_in[8];
  const float* woff  = (const float*)d_in[9];   // [192,256]
  const float* boff  = (const float*)d_in[10];
  const float* wattn = (const float*)d_in[11];  // [96,256]
  const float* battn = (const float*)d_in[12];
  const float* wval  = (const float*)d_in[13];  // [256,256]
  const float* bval  = (const float*)d_in[14];
  const float* wout  = (const float*)d_in[15];  // [256,256]
  const float* bout  = (const float*)d_in[16];
  const float* wff1  = (const float*)d_in[17];  // [2048,256]
  const float* bff1  = (const float*)d_in[18];
  const float* wff2  = (const float*)d_in[19];  // [256,2048]
  const float* bff2  = (const float*)d_in[20];
  const float* g1 = (const float*)d_in[21]; const float* be1 = (const float*)d_in[22];
  const float* g2 = (const float*)d_in[23]; const float* be2 = (const float*)d_in[24];
  const float* g3 = (const float*)d_in[25]; const float* be3 = (const float*)d_in[26];

  const int Bb = 2, Nq = 1024, C = 256, h = 8, R = Bb * Nq;  // R = 2048 rows
  const int HW0 = 200 * 176, HW2 = 400 * 352;

  // --- workspace carve-up ------------------------------------------------
  char* wsp = (char*)d_ws;
  auto alloc = [&](size_t bytes) {
    char* pp = wsp;
    wsp += (bytes + 255) & ~(size_t)255;
    return pp;
  };
  float* qkv   = (float*)alloc((size_t)R * 768 * 4);          // [R,768]
  float* sa    = (float*)alloc((size_t)R * C * 4);            // attn out
  float* sa2   = (float*)alloc((size_t)R * C * 4);            // out-proj
  float* q1    = (float*)alloc((size_t)R * C * 4);            // LN1
  float* offb  = (float*)alloc((size_t)R * 192 * 4);
  float* awb   = (float*)alloc((size_t)R * 96 * 4);
  float* ca_in = (float*)alloc((size_t)R * C * 4);
  float* ca    = (float*)alloc((size_t)R * C * 4);
  float* q2    = (float*)alloc((size_t)R * C * 4);
  float* ff2   = (float*)alloc((size_t)R * C * 4);
  // scores (64 MB) and FFN hidden (16 MB) are temporally disjoint -> union
  float* scores = (float*)alloc((size_t)16 * 1024 * 1024 * 4);
  float* ffh    = scores;
  unsigned short* val0 = (unsigned short*)alloc((size_t)Bb * HW0 * C * 2);
  unsigned short* val1 = (unsigned short*)alloc((size_t)Bb * HW0 * C * 2);
  unsigned short* val2 = (unsigned short*)alloc((size_t)Bb * HW2 * C * 2);

  auto gemm = [&](const float* A, const float* Bm, const float* bias, void* Cm,
                  int M, int N, int K,
                  long long sam, long long sak, long long sbk, long long sbn,
                  long long scm, long long scn,
                  long long ao, long long bo, long long co,
                  int z, int zdiv,
                  long long ab1, long long ab2, long long bb1, long long bb2,
                  long long cb1, long long cb2,
                  float alpha, int relu, int obf) {
    GemmParams p;
    p.A = A; p.B = Bm; p.bias = bias; p.C = Cm;
    p.M = M; p.N = N; p.K = K;
    p.sa_m = sam; p.sa_k = sak; p.sb_k = sbk; p.sb_n = sbn;
    p.sc_m = scm; p.sc_n = scn;
    p.aoff0 = ao; p.boff0 = bo; p.coff0 = co;
    p.zdiv = zdiv;
    p.a_b1 = ab1; p.a_b2 = ab2; p.b_b1 = bb1; p.b_b2 = bb2;
    p.c_b1 = cb1; p.c_b2 = cb2;
    p.alpha = alpha; p.relu = relu; p.out_bf16 = obf;
    wmma_gemm_bf16<<<dim3(M / BM, (N + BN - 1) / BN, z), TB, 0, stream>>>(p);
  };

  // ---- self-attention ---------------------------------------------------
  // qkv[b,n,:] = q[b,n,:] @ ipw^T + ipb  (A reads query transposed via strides)
  gemm(query, ipw, ipb, qkv, Nq, 768, C,
       /*A*/ 1, Nq, /*B*/ 1, C, /*C*/ 768, 1,
       0, 0, 0, Bb, 1, (long long)C * Nq, 0, 0, 0, (long long)Nq * 768, 0,
       1.0f, 0, 0);

  // scores[b,h] = (qh / sqrt(dh)) @ kh^T ; batch z = b*8 + head
  gemm(qkv, qkv, nullptr, scores, Nq, Nq, 32,
       /*A*/ 768, 1, /*B*/ 1, 768, /*C*/ Nq, 1,
       0, 256, 0, Bb * h, h,
       (long long)Nq * 768, 32, (long long)Nq * 768, 32,
       (long long)h * Nq * Nq, (long long)Nq * Nq,
       0.17677669529663687f /* 1/sqrt(32) */, 0, 0);

  row_softmax<<<Bb * h * Nq, TB, 0, stream>>>(scores, Nq);

  // sa[b,n,head*32+d] = attn @ vh
  gemm(scores, qkv, nullptr, sa, Nq, 32, Nq,
       /*A*/ Nq, 1, /*B*/ 768, 1, /*C*/ C, 1,
       0, 512, 0, Bb * h, h,
       (long long)h * Nq * Nq, (long long)Nq * Nq,
       (long long)Nq * 768, 32, (long long)Nq * C, 32,
       1.0f, 0, 0);

  // out-proj (B folded into M)
  gemm(sa, opw, opb, sa2, R, C, C, C, 1, 1, C, C, 1,
       0, 0, 0, 1, 1, 0, 0, 0, 0, 0, 0, 1.0f, 0, 0);

  {  // q1 = LN(q + sa2), residual reads query transposed
    LNParams p{sa2, query, g1, be1, q1, C,
               Nq, (long long)C * Nq, 1, Nq,   // residual: [B,C,N]
               1, C, 0, 1};                    // output: row-major
    ln_residual<<<R, TB, 0, stream>>>(p);
  }

  // ---- deformable cross-attention --------------------------------------
  // value projections (bf16 output) — the FLOP-dominant GEMMs
  const float* feats[3] = {bev, xc4, xc3};
  unsigned short* vouts[3] = {val0, val1, val2};
  const int HWs[3] = {HW0, HW0, HW2};
  for (int l = 0; l < 3; ++l) {
    gemm(feats[l], wval, bval, vouts[l], HWs[l], C, C,
         /*A*/ 1, HWs[l], /*B*/ 1, C, /*C*/ C, 1,
         0, 0, 0, Bb, 1, (long long)C * HWs[l], 0, 0, 0,
         (long long)HWs[l] * C, 0, 1.0f, 0, /*bf16*/ 1);
  }

  // offsets and attention-weight heads from q1
  gemm(q1, woff, boff, offb, R, 192, C, C, 1, 1, C, 192, 1,
       0, 0, 0, 1, 1, 0, 0, 0, 0, 0, 0, 1.0f, 0, 0);
  gemm(q1, wattn, battn, awb, R, 96, C, C, 1, 1, C, 96, 1,
       0, 0, 0, 1, 1, 0, 0, 0, 0, 0, 0, 1.0f, 0, 0);

  softmax12<<<(R * h + TB - 1) / TB, TB, 0, stream>>>(awb, R * h);

  deform_gather<<<(Bb * Nq * h * 32) / TB, TB, 0, stream>>>(
      qpos, offb, awb, val0, val1, val2, ca_in);

  gemm(ca_in, wout, bout, ca, R, C, C, C, 1, 1, C, C, 1,
       0, 0, 0, 1, 1, 0, 0, 0, 0, 0, 0, 1.0f, 0, 0);

  {  // q2 = LN(q1 + ca)
    LNParams p{ca, q1, g2, be2, q2, C, 1, C, 0, 1, 1, C, 0, 1};
    ln_residual<<<R, TB, 0, stream>>>(p);
  }

  // ---- FFN --------------------------------------------------------------
  gemm(q2, wff1, bff1, ffh, R, 2048, C, C, 1, 1, C, 2048, 1,
       0, 0, 0, 1, 1, 0, 0, 0, 0, 0, 0, 1.0f, /*relu*/ 1, 0);
  gemm(ffh, wff2, bff2, ff2, R, C, 2048, 2048, 1, 1, 2048, C, 1,
       0, 0, 0, 1, 1, 0, 0, 0, 0, 0, 0, 1.0f, 0, 0);

  {  // out = transpose(LN(q2 + ff2)) -> [B, C, N] written straight to d_out
    LNParams p{ff2, q2, g3, be3, (float*)d_out, C,
               1, C, 0, 1,
               Nq, (long long)C * Nq, 1, Nq};  // transposed output
    ln_residual<<<R, TB, 0, stream>>>(p);
  }
}
